// GATSMOTE_23433341567769
// MI455X (gfx1250) — matrix operand order repaired and checked by reading, not compiled, not run on hardware
//
#include <hip/hip_runtime.h>

typedef __bf16 bf16_t;
typedef __bf16 v16bf __attribute__((ext_vector_type(16)));
typedef __bf16 v8bf  __attribute__((ext_vector_type(8)));
typedef float  v8f   __attribute__((ext_vector_type(8)));

// ---------- helpers ----------
__device__ __forceinline__ bf16_t f2bf(float f) {
  unsigned u = __float_as_uint(f);
  u = u + 0x7FFFu + ((u >> 16) & 1u);          // round-to-nearest-even
  unsigned short h = (unsigned short)(u >> 16);
  return __builtin_bit_cast(bf16_t, h);
}

// monotonic float<->uint mapping so atomicMax(u32) == float max
__device__ __forceinline__ unsigned f_enc(float f) {
  unsigned b = __float_as_uint(f);
  return (b & 0x80000000u) ? ~b : (b | 0x80000000u);
}
__device__ __forceinline__ float f_dec(unsigned u) {
  unsigned b = (u & 0x80000000u) ? (u & 0x7FFFFFFFu) : ~u;
  return __uint_as_float(b);
}

// ---------- conversion / transpose ----------
__global__ void k_convert_bf16(const float* __restrict__ in, bf16_t* __restrict__ out, int n) {
  int i = blockIdx.x * blockDim.x + threadIdx.x;
  if (i < n) out[i] = f2bf(in[i]);
}

// W is [K, M] row-major f32 ; BT is [M, K] row-major bf16
__global__ void k_transpose_bf16(const float* __restrict__ W, bf16_t* __restrict__ BT,
                                 int K, int M) {
  int t = blockIdx.x * blockDim.x + threadIdx.x;
  if (t >= K * M) return;
  int m = t / K, k = t - m * K;
  BT[t] = f2bf(W[k * M + m]);
}

// ---------- WMMA GEMM: C[N,Mdim] = A[N,256] * B[256,Mdim], B given transposed ----------
// One wave32 per 16x16 output tile; K fixed to 256 (8 unrolled v_wmma_f32_16x16x32_bf16).
__global__ void __launch_bounds__(128)
k_wmma_gemm_bf16(const bf16_t* __restrict__ A, const bf16_t* __restrict__ BT,
                 float* __restrict__ C, int rowTiles, int Mdim) {
  const int K = 256;
  int wave = blockIdx.x * (blockDim.x >> 5) + (threadIdx.x >> 5);
  int mTiles = Mdim >> 4;
  if (wave >= rowTiles * mTiles) return;
  int tm = wave / mTiles;
  int tn = wave - tm * mTiles;
  int lane = threadIdx.x & 31;
  int lo = lane & 15, hi = lane >> 4;

  const bf16_t* arow = A  + (size_t)(tm * 16 + lo) * K;   // A fragment: row = lane&15
  const bf16_t* brow = BT + (size_t)(tn * 16 + lo) * K;   // B fragment: col = lane&15

  v8f acc = {};
#pragma unroll
  for (int k = 0; k < K; k += 32) {
    // A 16x32 bf16 layout: halves 0-7 -> K = k + hi*8 + 0..7 ; halves 8-15 -> +16
    v8bf a0 = *(const v8bf*)(arow + k + hi * 8);
    v8bf a1 = *(const v8bf*)(arow + k + 16 + hi * 8);
    v16bf av = __builtin_shufflevector(a0, a1,
        0, 1, 2, 3, 4, 5, 6, 7, 8, 9, 10, 11, 12, 13, 14, 15);
    // B 32x16 bf16 layout: halves 0-7 -> K = k + hi*8 + 0..7 ; halves 8-15 -> +16
    v8bf b0 = *(const v8bf*)(brow + k + hi * 8);
    v8bf b1 = *(const v8bf*)(brow + k + 16 + hi * 8);
    v16bf bv = __builtin_shufflevector(b0, b1,
        0, 1, 2, 3, 4, 5, 6, 7, 8, 9, 10, 11, 12, 13, 14, 15);
    acc = __builtin_amdgcn_wmma_f32_16x16x32_bf16(false, av, false, bv,
                                                  (short)0, acc, false, false);
  }
  // C/D layout: lanes 0-15 -> M = v ; lanes 16-31 -> M = v + 8 ; N = lane&15
  float* crow = C + (size_t)(tm * 16 + hi * 8) * Mdim + tn * 16 + lo;
#pragma unroll
  for (int v = 0; v < 8; ++v) crow[(size_t)v * Mdim] = acc[v];
}

// ---------- attention dot products: a_s/a_d [N,4] ----------
__global__ void k_attn(const float* __restrict__ h, const float* __restrict__ att_s,
                       const float* __restrict__ att_d, float* __restrict__ a_s,
                       float* __restrict__ a_d, int N) {
  int t = blockIdx.x * blockDim.x + threadIdx.x;
  if (t >= N * 4) return;
  int n = t >> 2, hd = t & 3;
  const float* hp  = h + (size_t)n * 256 + hd * 64;
  const float* asv = att_s + hd * 64;
  const float* adv = att_d + hd * 64;
  float ss = 0.f, sd = 0.f;
#pragma unroll
  for (int c = 0; c < 64; ++c) { float v = hp[c]; ss += v * asv[c]; sd += v * adv[c]; }
  a_s[t] = ss;
  a_d[t] = sd;
}

// ---------- zero accumulators ----------
__global__ void k_init(float* __restrict__ accum, unsigned* __restrict__ m_enc,
                       float* __restrict__ denom, int N) {
  int i = blockIdx.x * blockDim.x + threadIdx.x;
  if (i < N * 256) accum[i] = 0.f;
  if (i < N * 4) { m_enc[i] = 0u; denom[i] = 0.f; }
}

// ---------- edge pass 1: e = leaky_relu(a_s[src]+a_d[dst]); segment max ----------
__global__ void k_edge1(const int* __restrict__ ei, int E, int Et,
                        const float* __restrict__ a_s, const float* __restrict__ a_d,
                        float* __restrict__ e_store, unsigned* __restrict__ m_enc) {
  int t = blockIdx.x * blockDim.x + threadIdx.x;
  if (t >= Et * 4) return;
  int e = t >> 2, hd = t & 3;
  int s = (e < E) ? ei[e]     : (e - E);
  int d = (e < E) ? ei[E + e] : (e - E);
  float v = a_s[s * 4 + hd] + a_d[d * 4 + hd];
  v = (v > 0.f) ? v : 0.2f * v;                // LeakyReLU(0.2)
  e_store[t] = v;                              // t == e*4+hd
  atomicMax(m_enc + d * 4 + hd, f_enc(v));
}

// ---------- edge pass 2: ex = exp(e - m[dst]); segment sum ----------
__global__ void k_edge2(const int* __restrict__ ei, int E, int Et,
                        const unsigned* __restrict__ m_enc,
                        float* __restrict__ e_store, float* __restrict__ denom) {
  int t = blockIdx.x * blockDim.x + threadIdx.x;
  if (t >= Et * 4) return;
  int e = t >> 2, hd = t & 3;
  int d = (e < E) ? ei[E + e] : (e - E);
  float mm = f_dec(m_enc[d * 4 + hd]);
  float ex = __expf(e_store[t] - mm);
  e_store[t] = ex;
  atomicAdd(denom + d * 4 + hd, ex);
}

// ---------- edge pass 3: accum[dst] += h[src] * alpha  (one thread per edge-channel) ----------
__global__ void k_edge3(const int* __restrict__ ei, int E, long long Et,
                        const float* __restrict__ h, const float* __restrict__ e_store,
                        const float* __restrict__ denom, float* __restrict__ accum) {
  long long t = (long long)blockIdx.x * blockDim.x + threadIdx.x;
  if (t >= Et * 256) return;
  int e = (int)(t >> 8);
  int c = (int)(t & 255);
  int hd = c >> 6;
  int s = (e < E) ? ei[e]     : (e - E);
  int d = (e < E) ? ei[E + e] : (e - E);
  float alpha = e_store[e * 4 + hd] / (denom[d * 4 + hd] + 1e-16f);
  atomicAdd(accum + (size_t)d * 256 + c, h[(size_t)s * 256 + c] * alpha);
}

// ---------- bias (+relu) and convert to bf16 for next GEMM ----------
__global__ void k_bias_act_bf16(const float* __restrict__ acc, const float* __restrict__ bias,
                                bf16_t* __restrict__ out, int total, int do_relu) {
  int i = blockIdx.x * blockDim.x + threadIdx.x;
  if (i >= total) return;
  float v = acc[i] + bias[i & 255];
  if (do_relu) v = fmaxf(v, 0.f);
  out[i] = f2bf(v);
}

__global__ void k_bias16(float* __restrict__ out, const float* __restrict__ bc, int total) {
  int i = blockIdx.x * blockDim.x + threadIdx.x;
  if (i < total) out[i] += bc[i & 15];
}

// ---------- launcher ----------
extern "C" void kernel_launch(void* const* d_in, const int* in_sizes, int n_in,
                              void* d_out, int out_size, void* d_ws, size_t ws_size,
                              hipStream_t stream) {
  (void)n_in; (void)out_size; (void)ws_size;
  const float* x        = (const float*)d_in[0];
  const int*   ei       = (const int*)d_in[1];
  const float* W1       = (const float*)d_in[2];
  const float* att_src1 = (const float*)d_in[3];
  const float* att_dst1 = (const float*)d_in[4];
  const float* bias1    = (const float*)d_in[5];
  const float* W2       = (const float*)d_in[6];
  const float* att_src2 = (const float*)d_in[7];
  const float* att_dst2 = (const float*)d_in[8];
  const float* bias2    = (const float*)d_in[9];
  const float* Wc       = (const float*)d_in[10];
  const float* bc       = (const float*)d_in[11];

  const int N  = in_sizes[0] / 256;   // 50000 (divisible by 16)
  const int E  = in_sizes[1] / 2;     // 800000
  const int Et = E + N;               // edges + self-loops
  const int rowTiles = N / 16;

  // workspace carve-out (256B aligned slices)
  size_t off = 0;
  auto alloc = [&](size_t bytes) -> void* {
    void* r = (char*)d_ws + off;
    off += (bytes + 255) & ~(size_t)255;
    return r;
  };
  bf16_t*   xb      = (bf16_t*)alloc((size_t)N * 256 * sizeof(bf16_t));   // GEMM input
  bf16_t*   bt      = (bf16_t*)alloc((size_t)256 * 256 * sizeof(bf16_t)); // transposed weights
  float*    hbuf    = (float*)alloc((size_t)N * 256 * sizeof(float));     // h = x @ W
  float*    accum   = (float*)alloc((size_t)N * 256 * sizeof(float));     // segment-sum output
  float*    a_s     = (float*)alloc((size_t)N * 4 * sizeof(float));
  float*    a_d     = (float*)alloc((size_t)N * 4 * sizeof(float));
  unsigned* m_enc   = (unsigned*)alloc((size_t)N * 4 * sizeof(unsigned));
  float*    denom   = (float*)alloc((size_t)N * 4 * sizeof(float));
  float*    e_store = (float*)alloc((size_t)Et * 4 * sizeof(float));

  auto cdiv = [](long long a, long long b) { return (int)((a + b - 1) / b); };

  k_convert_bf16<<<cdiv((long long)N * 256, 256), 256, 0, stream>>>(x, xb, N * 256);

  auto gat_layer = [&](const float* W, const float* as_, const float* ad_,
                       const float* bias, int relu) {
    k_transpose_bf16<<<cdiv(256 * 256, 256), 256, 0, stream>>>(W, bt, 256, 256);
    k_wmma_gemm_bf16<<<cdiv((long long)rowTiles * 16, 4), 128, 0, stream>>>(
        xb, bt, hbuf, rowTiles, 256);
    k_attn<<<cdiv((long long)N * 4, 256), 256, 0, stream>>>(hbuf, as_, ad_, a_s, a_d, N);
    k_init<<<cdiv((long long)N * 256, 256), 256, 0, stream>>>(accum, m_enc, denom, N);
    k_edge1<<<cdiv((long long)Et * 4, 256), 256, 0, stream>>>(ei, E, Et, a_s, a_d,
                                                              e_store, m_enc);
    k_edge2<<<cdiv((long long)Et * 4, 256), 256, 0, stream>>>(ei, E, Et, m_enc,
                                                              e_store, denom);
    k_edge3<<<cdiv((long long)Et * 256, 256), 256, 0, stream>>>(ei, E, (long long)Et,
                                                                hbuf, e_store, denom, accum);
    k_bias_act_bf16<<<cdiv((long long)N * 256, 256), 256, 0, stream>>>(accum, bias, xb,
                                                                       N * 256, relu);
  };

  gat_layer(W1, att_src1, att_dst1, bias1, /*relu=*/1);
  gat_layer(W2, att_src2, att_dst2, bias2, /*relu=*/0);

  // classifier: logits[N,16] = out2_bf16 @ Wc + bc
  k_transpose_bf16<<<cdiv(256 * 16, 256), 256, 0, stream>>>(Wc, bt, 256, 16);
  k_wmma_gemm_bf16<<<cdiv((long long)rowTiles, 4), 128, 0, stream>>>(
      xb, bt, (float*)d_out, rowTiles, 16);
  k_bias16<<<cdiv((long long)N * 16, 256), 256, 0, stream>>>((float*)d_out, bc, N * 16);
}
